// GraphQNNHybrid_65481071399768
// MI455X (gfx1250) — matrix operand order repaired and checked by reading, not compiled, not run on hardware
//
#include <hip/hip_runtime.h>
#include <hip/hip_bf16.h>
#include <math.h>

// Problem sizes (fixed by the reference).
#define DD    4096   // layer width
#define BATCH 4096   // batch columns

typedef __attribute__((ext_vector_type(16))) __bf16 v16bf;
typedef __attribute__((ext_vector_type(8)))  float  v8f;

// ---------------- GEMM tile configuration ----------------
// Block: 256 threads = 8 wave32, waves arranged 4 (M) x 2 (N).
// Block tile 256(M) x 128(N), K-slab 64. Per-wave tile 64x64 = 4x4 WMMA tiles.
// LDS double-buffered: async loads for slab k+1 overlap WMMA compute on slab k.
#define BM 256
#define BN 128
#define BK 64
#define LDA_P 72   // padded LDS row stride (elements) = BK + 8
#define LDB_P 72

// One 16-byte per-lane async copy global -> LDS (CDNA5 GLOBAL_LOAD_ASYNC_TO_LDS_B128).
__device__ __forceinline__ void async_cp16(unsigned int lds_byte_addr, const void* gaddr) {
  asm volatile("global_load_async_to_lds_b128 %0, %1, off"
               :: "v"(lds_byte_addr), "v"(gaddr) : "memory");
}

__device__ __forceinline__ unsigned int lds_addr(const void* p) {
  // Low 32 bits of a generic pointer to __shared__ data are the LDS byte address.
  return (unsigned int)(size_t)p;
}

// C = tanh(W @ H), with H kept transposed: Hin/Hout are (BATCH x DD) bf16 row-major,
// Hin[n][k] == H[k][n]. W is (DD x DD) bf16 row-major.
__global__ __launch_bounds__(256)
void gemm_tanh_bf16(const unsigned short* __restrict__ Wm,
                    const unsigned short* __restrict__ Hin,
                    unsigned short* __restrict__ Hout) {
  __shared__ unsigned short sA[2][BM * LDA_P];   // 2 x 36 KB
  __shared__ unsigned short sB[2][BN * LDB_P];   // 2 x 18 KB

  const __bf16* Wb = (const __bf16*)Wm;
  const __bf16* Hb = (const __bf16*)Hin;
  __bf16*       Ho = (__bf16*)Hout;

  const int tid  = threadIdx.x;
  const int lane = tid & 31;
  const int wave = tid >> 5;
  const int wm   = wave & 3;     // wave row (M)
  const int wn   = wave >> 2;    // wave col (N)
  const int half = lane >> 4;    // 0: lanes 0-15, 1: lanes 16-31
  const int l16  = lane & 15;

  const int m_blk = blockIdx.y * BM;
  const int n_blk = blockIdx.x * BN;

  const int ldr = tid >> 3;        // 0..31 : row within a load pass
  const int ldc = (tid & 7) * 8;   // element column of the 16B chunk

  v8f acc[4][4] = {};

  // Issue the async stage for one K-slab into LDS buffer `buf` (no wait).
  auto stage = [&](int buf, int k0) {
#pragma unroll
    for (int p = 0; p < 8; ++p) {           // A tile: BM x BK
      int row = ldr + p * 32;
      async_cp16(lds_addr(&sA[buf][row * LDA_P + ldc]),
                 Wb + (size_t)(m_blk + row) * DD + k0 + ldc);
    }
#pragma unroll
    for (int p = 0; p < 4; ++p) {           // B tile: BN x BK
      int row = ldr + p * 32;
      async_cp16(lds_addr(&sB[buf][row * LDB_P + ldc]),
                 Hb + (size_t)(n_blk + row) * DD + k0 + ldc);
    }
  };

  // Prologue: fill buffer 0.
  stage(0, 0);
  __builtin_amdgcn_s_wait_asynccnt(0);
  __syncthreads();

  for (int k0 = 0; k0 < DD; k0 += BK) {
    const int cur = (k0 >> 6) & 1;

    // Prefetch next slab into the other buffer while we compute on this one.
    if (k0 + BK < DD) stage(cur ^ 1, k0 + BK);

#pragma unroll
    for (int kk = 0; kk < BK; kk += 32) {
      union { uint4 q[2]; v16bf v; } a[4], b[4];
      // A fragment (16x32 bf16): lane half=0 holds K 0..7 & 16..23, half=1 holds 8..15 & 24..31.
#pragma unroll
      for (int i = 0; i < 4; ++i) {
        const unsigned short* base = &sA[cur][(wm * 64 + i * 16 + l16) * LDA_P + kk + half * 8];
        a[i].q[0] = *(const uint4*)(base);
        a[i].q[1] = *(const uint4*)(base + 16);   // +16 elements = K+16
      }
      // B fragment (32x16 bf16): lane half=0 holds K 0..7 & 16..23, half=1 holds 8..15 & 24..31.
#pragma unroll
      for (int j = 0; j < 4; ++j) {
        const unsigned short* base = &sB[cur][(wn * 64 + j * 16 + l16) * LDB_P + kk + half * 8];
        b[j].q[0] = *(const uint4*)(base);
        b[j].q[1] = *(const uint4*)(base + 16);
      }
#pragma unroll
      for (int i = 0; i < 4; ++i)
#pragma unroll
        for (int j = 0; j < 4; ++j)
          acc[i][j] = __builtin_amdgcn_wmma_f32_16x16x32_bf16(
              false, a[i].v, false, b[j].v, (short)0, acc[i][j], false, false);
    }

    // Drain our prefetch, then one barrier: everyone is done reading `cur`
    // and everyone's async writes to the other buffer have landed.
    __builtin_amdgcn_s_wait_asynccnt(0);
    __syncthreads();
  }

  // Epilogue: tanh, convert to bf16, store transposed (contiguous 16B per lane).
#pragma unroll
  for (int i = 0; i < 4; ++i) {
#pragma unroll
    for (int j = 0; j < 4; ++j) {
      union { v8f v; float f[8]; } c; c.v = acc[i][j];
      union { __bf16 h[8]; uint4 q; } o;
#pragma unroll
      for (int g = 0; g < 8; ++g) o.h[g] = (__bf16)tanhf(c.f[g]);
      int n = n_blk + wn * 64 + j * 16 + l16;
      int m = m_blk + wm * 64 + i * 16 + half * 8;   // 8 contiguous rows of C = contiguous in Ht
      *(uint4*)(Ho + (size_t)n * DD + m) = o.q;
    }
  }
}

// float32 (DD x DD) -> bf16, elementwise (vectorized by 4).
__global__ void cvt_f32_to_bf16(const float* __restrict__ src, unsigned short* __restrict__ dst) {
  size_t i = (size_t)blockIdx.x * blockDim.x + threadIdx.x;
  float4 v = ((const float4*)src)[i];
  union { __bf16 h[4]; uint2 u; } o;
  o.h[0] = (__bf16)v.x; o.h[1] = (__bf16)v.y; o.h[2] = (__bf16)v.z; o.h[3] = (__bf16)v.w;
  ((uint2*)dst)[i] = o.u;
}

// x (DD x BATCH f32 row-major) -> Ht0 (BATCH x DD bf16 row-major), transposed via LDS tile.
__global__ void transpose_x_to_bf16(const float* __restrict__ X, unsigned short* __restrict__ Ht) {
  __shared__ float tile[32][33];
  const int b0 = blockIdx.x * 32;
  const int d0 = blockIdx.y * 32;
  const int tx = threadIdx.x, ty = threadIdx.y;   // block (32, 8)
  __bf16* H = (__bf16*)Ht;
  for (int i = ty; i < 32; i += 8)
    tile[i][tx] = X[(size_t)(d0 + i) * BATCH + b0 + tx];
  __syncthreads();
  for (int i = ty; i < 32; i += 8)
    H[(size_t)(b0 + i) * DD + d0 + tx] = (__bf16)tile[tx][i];
}

// Stage 1 of batch-mean: partial[pb][d] = sum over 128 rows of Ht[b][d].
__global__ void colsum_partial(const unsigned short* __restrict__ Ht, float* __restrict__ partial) {
  const __bf16* H = (const __bf16*)Ht;
  int d  = blockIdx.x * 256 + threadIdx.x;
  int pb = blockIdx.y;                       // 0..31, each covers 128 rows
  float s = 0.f;
  for (int r = 0; r < 128; ++r)
    s += (float)H[(size_t)(pb * 128 + r) * DD + d];
  partial[(size_t)pb * DD + d] = s;
}

// Head: q = cos(mean), logits = [q, 1-q], softmax over 8192. One 1024-thread block.
__global__ __launch_bounds__(1024)
void head_kernel(const float* __restrict__ partial, float* __restrict__ out) {
  __shared__ float red[1024];
  const int tid = threadIdx.x;
  float q[4];
#pragma unroll
  for (int t = 0; t < 4; ++t) {
    int d = tid * 4 + t;
    float s = 0.f;
    for (int p = 0; p < 32; ++p) s += partial[(size_t)p * DD + d];
    q[t] = cosf(s * (1.0f / (float)BATCH));
  }
  float mx = -1e30f;
#pragma unroll
  for (int t = 0; t < 4; ++t) mx = fmaxf(mx, fmaxf(q[t], 1.0f - q[t]));
  red[tid] = mx; __syncthreads();
  for (int s = 512; s > 0; s >>= 1) {
    if (tid < s) red[tid] = fmaxf(red[tid], red[tid + s]);
    __syncthreads();
  }
  mx = red[0]; __syncthreads();
  float se = 0.f;
#pragma unroll
  for (int t = 0; t < 4; ++t) se += expf(q[t] - mx) + expf(1.0f - q[t] - mx);
  red[tid] = se; __syncthreads();
  for (int s = 512; s > 0; s >>= 1) {
    if (tid < s) red[tid] += red[tid + s];
    __syncthreads();
  }
  float inv = 1.0f / red[0];
#pragma unroll
  for (int t = 0; t < 4; ++t) {
    int d = tid * 4 + t;
    out[d]      = expf(q[t] - mx) * inv;
    out[DD + d] = expf(1.0f - q[t] - mx) * inv;
  }
}

extern "C" void kernel_launch(void* const* d_in, const int* in_sizes, int n_in,
                              void* d_out, int out_size, void* d_ws, size_t ws_size,
                              hipStream_t stream) {
  (void)in_sizes; (void)n_in; (void)out_size; (void)ws_size;
  const float* x  = (const float*)d_in[0];
  const float* W0 = (const float*)d_in[1];
  const float* W1 = (const float*)d_in[2];
  const float* W2 = (const float*)d_in[3];
  float* out = (float*)d_out;

  // Workspace layout (~168.3 MB):
  //   3 bf16 weight copies (32 MB each), 2 bf16 activation ping-pong (32 MB each),
  //   32 x DD float partial column sums (512 KB).
  char* ws = (char*)d_ws;
  const size_t WBYTES = (size_t)DD * DD * 2;       // 32 MB bf16
  unsigned short* wb0 = (unsigned short*)(ws);
  unsigned short* wb1 = (unsigned short*)(ws + WBYTES);
  unsigned short* wb2 = (unsigned short*)(ws + 2 * WBYTES);
  unsigned short* hA  = (unsigned short*)(ws + 3 * WBYTES);
  unsigned short* hB  = (unsigned short*)(ws + 4 * WBYTES);
  float* partial      = (float*)(ws + 5 * WBYTES);

  const int nvec = DD * DD / 4;   // float4 elements per weight matrix
  cvt_f32_to_bf16<<<nvec / 256, 256, 0, stream>>>(W0, wb0);
  cvt_f32_to_bf16<<<nvec / 256, 256, 0, stream>>>(W1, wb1);
  cvt_f32_to_bf16<<<nvec / 256, 256, 0, stream>>>(W2, wb2);

  transpose_x_to_bf16<<<dim3(BATCH / 32, DD / 32), dim3(32, 8), 0, stream>>>(x, hA);

  dim3 gg(BATCH / BN, DD / BM);   // (32, 16)
  gemm_tanh_bf16<<<gg, 256, 0, stream>>>(wb0, hA, hB);
  gemm_tanh_bf16<<<gg, 256, 0, stream>>>(wb1, hB, hA);
  gemm_tanh_bf16<<<gg, 256, 0, stream>>>(wb2, hA, hB);

  colsum_partial<<<dim3(DD / 256, 32), 256, 0, stream>>>(hB, partial);
  head_kernel<<<1, 1024, 0, stream>>>(partial, out);
}
